// TransformerEncoderBlock_7842610283012
// MI455X (gfx1250) — compile-verified, matrix operand order checked by reading, not run on hardware
//
#include <hip/hip_runtime.h>
#include <hip/hip_bf16.h>
#include <math.h>

// ---------------------------------------------------------------------------
// Transformer encoder block for MI455X (gfx1250), wave32, WMMA f16->f32.
//
// Shapes: B=4, S=2048, E=768, H=12, D=64, FF=3072, M = B*S = 8192.
// All GEMMs run on v_wmma_f32_16x16x32_f16 (f16 A/B, f32 accum).
// attn_weights (805 MB fp32) dominates memory traffic:
//   - fused score-GEMM + softmax writes it once,
//   - attention is processed in 4-head chunks (67 MB P slice << 192 MB L2)
//     so the P x V pass reads P from L2, not HBM,
//   - V tiles stream to LDS via CDNA5 async copies (ASYNCcnt), double-buffered.
// ---------------------------------------------------------------------------

typedef __attribute__((ext_vector_type(16))) _Float16 v16h;
typedef __attribute__((ext_vector_type(8)))  _Float16 v8h;
typedef __attribute__((ext_vector_type(8)))  float    v8f;

#define EMBED 768
#define SEQ   2048
#define BATCH 4
#define HEADS 12
#define HDIM  64
#define FFDIM 3072
#define MROWS (BATCH * SEQ)           // 8192
#define OUTEL ((size_t)MROWS * EMBED) // 6291456

__device__ __forceinline__ v8f wmma_f16(v16h a, v16h b, v8f c) {
    // D = A(16x32 f16) x B(32x16 f16) + C(16x16 f32)
    return __builtin_amdgcn_wmma_f32_16x16x32_f16(
        false, a, false, b, (short)0, c, false, false);
}

// Load a 16x32 f16 A-fragment (or B-fragment of a [N,K] K-contiguous matrix).
// Per ISA layout: lane half=l/16; VGPRs 0-3 hold K = half*8 + 0..7,
// VGPRs 4-7 hold K = 16 + half*8 + 0..7.  base = row pointer at k0.
__device__ __forceinline__ v16h load_frag(const _Float16* base, int half) {
    v8h lo = *(const v8h*)(base + half * 8);
    v8h hi = *(const v8h*)(base + 16 + half * 8);
    v16h r;
#pragma unroll
    for (int i = 0; i < 8; ++i) { r[i] = lo[i]; r[i + 8] = hi[i]; }
    return r;
}

// --------------------------- weight convert+transpose ----------------------
// in: f32 row-major [K,N]  ->  out: f16 [N,K] (K contiguous)
__global__ void convert_transpose(const float* __restrict__ in,
                                  _Float16* __restrict__ out, int K, int N) {
    size_t i = (size_t)blockIdx.x * 256 + threadIdx.x;
    if (i < (size_t)K * N) {
        int k = (int)(i / N);
        int n = (int)(i % N);
        out[(size_t)n * K + k] = (_Float16)in[i];
    }
}

// --------------------------- layernorm -> f16 -------------------------------
__global__ void layernorm_f16(const float* __restrict__ x,
                              const float* __restrict__ w,
                              const float* __restrict__ b,
                              _Float16* __restrict__ out) {
    __shared__ float s1[256], s2[256];
    int row = blockIdx.x;
    int t = threadIdx.x;
    const float* xr = x + (size_t)row * EMBED;
    float v0 = xr[t], v1 = xr[t + 256], v2 = xr[t + 512];
    s1[t] = v0 + v1 + v2;
    s2[t] = v0 * v0 + v1 * v1 + v2 * v2;
    __syncthreads();
    for (int off = 128; off; off >>= 1) {
        if (t < off) { s1[t] += s1[t + off]; s2[t] += s2[t + off]; }
        __syncthreads();
    }
    float mean = s1[0] * (1.0f / EMBED);
    float var  = s2[0] * (1.0f / EMBED) - mean * mean;
    float inv  = rsqrtf(var + 1e-5f);
    _Float16* o = out + (size_t)row * EMBED;
    o[t]       = (_Float16)((v0 - mean) * inv * w[t]       + b[t]);
    o[t + 256] = (_Float16)((v1 - mean) * inv * w[t + 256] + b[t + 256]);
    o[t + 512] = (_Float16)((v2 - mean) * inv * w[t + 512] + b[t + 512]);
}

// --------------------------- generic WMMA GEMM ------------------------------
// C[M,N] = epi( A16[M,K] @ Bt16[N,K]^T + bias[N] )
// Block: 256 thr = 8 waves; macro tile 64(M) x 128(N); wave = 16x64 (4 tiles).
// Fragments load straight from global: weights are L2-resident (192 MB L2).
// EPI: 0 = f16 out, 1 = f32 out, 2 = f16 out + exact GELU, 3 = f32 + residual
template <int EPI>
__global__ void gemm_wmma(const _Float16* __restrict__ A,
                          const _Float16* __restrict__ Bt,
                          const float* __restrict__ bias,
                          const float* __restrict__ res,
                          float* __restrict__ Cf,
                          _Float16* __restrict__ Ch,
                          int M, int N, int K) {
    int m0 = blockIdx.x * 64;
    int n0 = blockIdx.y * 128;
    int tid = threadIdx.x;
    int wave = tid >> 5, lane = tid & 31, half = lane >> 4, lm = lane & 15;
    int wm = wave & 3, wn = wave >> 2;

    const _Float16* arow = A + (size_t)(m0 + wm * 16 + lm) * K;
    v8f acc[4] = {};

    for (int k0 = 0; k0 < K; k0 += 32) {
        v16h a = load_frag(arow + k0, half);
#pragma unroll
        for (int nt = 0; nt < 4; ++nt) {
            const _Float16* brow =
                Bt + (size_t)(n0 + wn * 64 + nt * 16 + lm) * K + k0;
            v16h bf = load_frag(brow, half);
            acc[nt] = wmma_f16(a, bf, acc[nt]);
        }
    }

#pragma unroll
    for (int nt = 0; nt < 4; ++nt) {
        int col = n0 + wn * 64 + nt * 16 + lm;
        float bv = bias[col];
#pragma unroll
        for (int r = 0; r < 8; ++r) {
            int row = m0 + wm * 16 + 8 * half + r;
            float v = acc[nt][r] + bv;
            size_t idx = (size_t)row * N + col;
            if (EPI == 2) v = 0.5f * v * (1.0f + erff(v * 0.70710678f));
            if (EPI == 0 || EPI == 2) {
                Ch[idx] = (_Float16)v;
            } else {
                if (EPI == 3) v += res[idx];
                Cf[idx] = v;
            }
        }
    }
}

// ----------------- fused attention scores + softmax --------------------------
// One block per (head-in-chunk, 16-row slab). LDS: 16 x 2048 f32 (128 KB).
// 8 waves each own a 256-col slab: 16 N-tiles, K=64 (two k-steps per tile).
__global__ void attn_scores_softmax(const _Float16* __restrict__ q16,
                                    const _Float16* __restrict__ k16,
                                    float* __restrict__ P, int bh0) {
    extern __shared__ float srow[]; // [16][2048]
    int mblk = blockIdx.x & 127;
    int bh = bh0 + (blockIdx.x >> 7);
    int h = bh % HEADS, b = bh / HEADS;
    int tid = threadIdx.x;
    int wave = tid >> 5, lane = tid & 31, half = lane >> 4, lm = lane & 15;

    const _Float16* Qb = q16 + ((size_t)(b * SEQ + mblk * 16)) * EMBED + h * HDIM;
    const _Float16* Kb = k16 + ((size_t)(b * SEQ)) * EMBED + h * HDIM;

    const _Float16* arow = Qb + (size_t)lm * EMBED;
    v16h a0 = load_frag(arow, half);
    v16h a1 = load_frag(arow + 32, half);

    for (int nt = 0; nt < 16; ++nt) {
        int n0 = wave * 256 + nt * 16;
        const _Float16* brow = Kb + (size_t)(n0 + lm) * EMBED;
        v16h b0 = load_frag(brow, half);
        v16h b1 = load_frag(brow + 32, half);
        v8f acc = {};
        acc = wmma_f16(a0, b0, acc);
        acc = wmma_f16(a1, b1, acc);
#pragma unroll
        for (int r = 0; r < 8; ++r)
            srow[(size_t)(r + 8 * half) * SEQ + n0 + lm] = acc[r] * 0.125f;
    }
    __syncthreads();

    // softmax: wave w handles rows w and w+8
    for (int rr = 0; rr < 2; ++rr) {
        int row = wave + rr * 8;
        float* s = srow + (size_t)row * SEQ;
        float mx = -1e30f;
        for (int c = lane; c < SEQ; c += 32) mx = fmaxf(mx, s[c]);
        for (int off = 16; off; off >>= 1) mx = fmaxf(mx, __shfl_xor(mx, off, 32));
        float sum = 0.0f;
        for (int c = lane; c < SEQ; c += 32) {
            float e = expf(s[c] - mx);
            s[c] = e;
            sum += e;
        }
        for (int off = 16; off; off >>= 1) sum += __shfl_xor(sum, off, 32);
        float inv = 1.0f / sum;
        float* prow = P + (((size_t)(b * HEADS + h)) * SEQ + mblk * 16 + row) * SEQ;
        for (int c = lane; c < SEQ; c += 32) prow[c] = s[c] * inv;
    }
}

// ----------------- attention P x V (async double-buffered V tiles) ----------
// One block per (head-in-chunk, 64-row slab); 128 thr = 4 waves, wave = 16x64.
// V tile [32k x 64n] streamed global->LDS with GLOBAL_LOAD_ASYNC_TO_LDS_B128
// (ASYNCcnt), double buffered so tile i+1 copies while tile i feeds WMMAs.
// P tiles (fresh in L2 from the score pass) convert f32->f16 in registers.
__global__ void attn_pv(const float* __restrict__ P,
                        const _Float16* __restrict__ vmat,
                        _Float16* __restrict__ attn16, int bh0) {
    __shared__ _Float16 Vraw[2 * 32 * 64]; // two row-major [k][n] tiles (8 KB)
    int bh = bh0 + blockIdx.x;
    int h = bh % HEADS, b = bh / HEADS;
    int mblk = blockIdx.y; // 32 slabs of 64 rows
    int tid = threadIdx.x;
    int wave = tid >> 5, lane = tid & 31, half = lane >> 4, lm = lane & 15;

    const float* Pb = P + (((size_t)(b * HEADS + h)) * SEQ + mblk * 64) * SEQ;
    const _Float16* Vb = vmat + ((size_t)(b * SEQ)) * EMBED + h * HDIM;

    v8f acc[4] = {};
    const float* prow = Pb + (size_t)(wave * 16 + lm) * SEQ;
    unsigned ldsbase = (unsigned)(uintptr_t)(&Vraw[0]); // flat addr[31:0] = LDS

    // each thread async-copies two 16-byte chunks per tile (128 thr * 32 B)
    auto issue_tile = [&](int k0, int bufidx) {
#pragma unroll
        for (int c = 0; c < 2; ++c) {
            int chunk = tid * 2 + c;     // 0..255
            int kk = chunk >> 3;         // 0..31
            int n8 = (chunk & 7) * 8;    // 0..56
            unsigned lds = ldsbase + (unsigned)(bufidx * 4096 + (kk * 64 + n8) * 2);
            unsigned long long ga =
                (unsigned long long)(uintptr_t)(Vb + (size_t)(k0 + kk) * EMBED + n8);
            asm volatile("global_load_async_to_lds_b128 %0, %1, off"
                         :: "v"(lds), "v"(ga) : "memory");
        }
    };

    const int NIT = SEQ / 32; // 64
    issue_tile(0, 0);
    for (int it = 0; it < NIT; ++it) {
        if (it + 1 < NIT) {
            issue_tile((it + 1) * 32, (it + 1) & 1);
            asm volatile("s_wait_asynccnt 0x2" ::: "memory"); // tile `it` landed
        } else {
            asm volatile("s_wait_asynccnt 0x0" ::: "memory");
        }
        __syncthreads(); // tile `it` visible from all waves

        // A fragment: f32 P -> f16
        v16h a;
        const float* pk = prow + it * 32;
#pragma unroll
        for (int i = 0; i < 8; ++i) {
            a[i]     = (_Float16)pk[half * 8 + i];
            a[i + 8] = (_Float16)pk[16 + half * 8 + i];
        }
        const _Float16* vb = Vraw + (it & 1) * 2048; // [k][n], k-major
#pragma unroll
        for (int nt = 0; nt < 4; ++nt) {
            int n = nt * 16 + lm;
            v16h bf;
#pragma unroll
            for (int i = 0; i < 8; ++i) {
                bf[i]     = vb[(half * 8 + i) * 64 + n];
                bf[i + 8] = vb[(16 + half * 8 + i) * 64 + n];
            }
            acc[nt] = wmma_f16(a, bf, acc[nt]);
        }
        __syncthreads(); // tile `it` consumed; its buffer reusable at it+2
    }

    size_t rowbase = (size_t)b * SEQ + mblk * 64 + wave * 16;
#pragma unroll
    for (int nt = 0; nt < 4; ++nt) {
        int col = h * HDIM + nt * 16 + lm;
#pragma unroll
        for (int r = 0; r < 8; ++r)
            attn16[(rowbase + r + 8 * half) * EMBED + col] = (_Float16)acc[nt][r];
    }
}

// ----------------- sigmoid gate combine -------------------------------------
__global__ void gate_combine(const float* __restrict__ x,
                             const float* __restrict__ O,
                             const float* __restrict__ G,
                             const float* __restrict__ Sk,
                             float* __restrict__ hout, size_t n) {
    size_t i = (size_t)blockIdx.x * 256 + threadIdx.x;
    if (i < n) {
        float g = 1.0f / (1.0f + expf(-G[i]));
        hout[i] = x[i] + g * O[i] + (1.0f - g) * Sk[i];
    }
}

// ---------------------------------------------------------------------------
extern "C" void kernel_launch(void* const* d_in, const int* in_sizes, int n_in,
                              void* d_out, int out_size, void* d_ws, size_t ws_size,
                              hipStream_t stream) {
    const float* x      = (const float*)d_in[0];
    const float* ln1_w  = (const float*)d_in[1];
    const float* ln1_b  = (const float*)d_in[2];
    const float* w_q    = (const float*)d_in[3];
    const float* w_k    = (const float*)d_in[4];
    const float* w_v    = (const float*)d_in[5];
    const float* w_o    = (const float*)d_in[6];
    const float* b_q    = (const float*)d_in[7];
    const float* b_k    = (const float*)d_in[8];
    const float* b_v    = (const float*)d_in[9];
    const float* b_o    = (const float*)d_in[10];
    const float* w_gate = (const float*)d_in[11];
    const float* b_gate = (const float*)d_in[12];
    const float* w_skip = (const float*)d_in[13];
    const float* b_skip = (const float*)d_in[14];
    const float* ln2_w  = (const float*)d_in[15];
    const float* ln2_b  = (const float*)d_in[16];
    const float* w1     = (const float*)d_in[17];
    const float* b1     = (const float*)d_in[18];
    const float* w2     = (const float*)d_in[19];
    const float* b2     = (const float*)d_in[20];

    float* out = (float*)d_out;
    float* P   = out + OUTEL; // attn_weights region [B,H,S,S]

    // ---- workspace carve-out ----
    char* p = (char*)d_ws;
    auto alloc = [&](size_t bytes) -> void* {
        void* r = (void*)p;
        p += (bytes + 255) & ~(size_t)255;
        return r;
    };
    const size_t EE = (size_t)EMBED * EMBED;
    _Float16* wqt = (_Float16*)alloc(EE * 2);
    _Float16* wkt = (_Float16*)alloc(EE * 2);
    _Float16* wvt = (_Float16*)alloc(EE * 2);
    _Float16* wot = (_Float16*)alloc(EE * 2);
    _Float16* wgt = (_Float16*)alloc(EE * 2);
    _Float16* wst = (_Float16*)alloc(EE * 2);
    _Float16* w1t = (_Float16*)alloc((size_t)EMBED * FFDIM * 2); // [FF, E]
    _Float16* w2t = (_Float16*)alloc((size_t)FFDIM * EMBED * 2); // [E, FF]
    _Float16* xn16   = (_Float16*)alloc(OUTEL * 2);
    _Float16* q16    = (_Float16*)alloc(OUTEL * 2);
    _Float16* k16    = (_Float16*)alloc(OUTEL * 2);
    _Float16* v16p   = (_Float16*)alloc(OUTEL * 2);
    _Float16* attn16 = (_Float16*)alloc(OUTEL * 2);
    _Float16* hn16   = (_Float16*)alloc(OUTEL * 2);
    _Float16* ff1    = (_Float16*)alloc((size_t)MROWS * FFDIM * 2);
    float* Of  = (float*)alloc(OUTEL * 4);
    float* Gf  = (float*)alloc(OUTEL * 4);
    float* Skf = (float*)alloc(OUTEL * 4);
    float* hf  = (float*)alloc(OUTEL * 4);

    // ---- 1. weight convert + transpose to f16 [N,K] ----
    int g_ee = (int)((EE + 255) / 256);
    convert_transpose<<<g_ee, 256, 0, stream>>>(w_q, wqt, EMBED, EMBED);
    convert_transpose<<<g_ee, 256, 0, stream>>>(w_k, wkt, EMBED, EMBED);
    convert_transpose<<<g_ee, 256, 0, stream>>>(w_v, wvt, EMBED, EMBED);
    convert_transpose<<<g_ee, 256, 0, stream>>>(w_o, wot, EMBED, EMBED);
    convert_transpose<<<g_ee, 256, 0, stream>>>(w_gate, wgt, EMBED, EMBED);
    convert_transpose<<<g_ee, 256, 0, stream>>>(w_skip, wst, EMBED, EMBED);
    int g_ef = (int)(((size_t)EMBED * FFDIM + 255) / 256);
    convert_transpose<<<g_ef, 256, 0, stream>>>(w1, w1t, EMBED, FFDIM);
    convert_transpose<<<g_ef, 256, 0, stream>>>(w2, w2t, FFDIM, EMBED);

    // ---- 2. LN1 ----
    layernorm_f16<<<MROWS, 256, 0, stream>>>(x, ln1_w, ln1_b, xn16);

    // ---- 3. Q/K/V projections (f16 out) ----
    dim3 gE(MROWS / 64, EMBED / 128);
    gemm_wmma<0><<<gE, 256, 0, stream>>>(xn16, wqt, b_q, nullptr, nullptr, q16,
                                         MROWS, EMBED, EMBED);
    gemm_wmma<0><<<gE, 256, 0, stream>>>(xn16, wkt, b_k, nullptr, nullptr, k16,
                                         MROWS, EMBED, EMBED);
    gemm_wmma<0><<<gE, 256, 0, stream>>>(xn16, wvt, b_v, nullptr, nullptr, v16p,
                                         MROWS, EMBED, EMBED);

    // ---- 4+5. attention in 4-head chunks: P slice (67 MB) stays L2-resident
    //           between the score/softmax pass and the P x V pass.
    const int HPC = 4; // heads per chunk
    for (int bh0 = 0; bh0 < BATCH * HEADS; bh0 += HPC) {
        attn_scores_softmax<<<HPC * (SEQ / 16), 256, 16 * SEQ * sizeof(float),
                              stream>>>(q16, k16, P, bh0);
        attn_pv<<<dim3(HPC, SEQ / 64), 128, 0, stream>>>(P, v16p, attn16, bh0);
    }

    // ---- 6. O / gate / skip projections (f32 out) ----
    gemm_wmma<1><<<gE, 256, 0, stream>>>(attn16, wot, b_o, nullptr, Of, nullptr,
                                         MROWS, EMBED, EMBED);
    gemm_wmma<1><<<gE, 256, 0, stream>>>(xn16, wgt, b_gate, nullptr, Gf, nullptr,
                                         MROWS, EMBED, EMBED);
    gemm_wmma<1><<<gE, 256, 0, stream>>>(xn16, wst, b_skip, nullptr, Skf, nullptr,
                                         MROWS, EMBED, EMBED);

    // ---- 7. gate combine + residual -> h ----
    gate_combine<<<(int)((OUTEL + 255) / 256), 256, 0, stream>>>(x, Of, Gf, Skf,
                                                                 hf, OUTEL);

    // ---- 8. LN2 ----
    layernorm_f16<<<MROWS, 256, 0, stream>>>(hf, ln2_w, ln2_b, hn16);

    // ---- 9. FFN ----
    dim3 gF(MROWS / 64, FFDIM / 128);
    gemm_wmma<2><<<gF, 256, 0, stream>>>(hn16, w1t, b1, nullptr, nullptr, ff1,
                                         MROWS, FFDIM, EMBED);
    gemm_wmma<3><<<gE, 256, 0, stream>>>(ff1, w2t, b2, hf, out, nullptr,
                                         MROWS, EMBED, FFDIM);
}
